// HypAttAgg_541165879723
// MI455X (gfx1250) — compile-verified, matrix operand order checked by reading, not run on hardware
//
#include <hip/hip_runtime.h>

#define NN 50000
#define NE 800000
#define DD 64

typedef __attribute__((ext_vector_type(16))) _Float16 v16h;
typedef __attribute__((ext_vector_type(8)))  float    v8f;

// Monotonic float -> uint encoding so unsigned atomicMax == float max.
__device__ __forceinline__ unsigned enc_f32(float f) {
    unsigned u = __float_as_uint(f);
    return (u & 0x80000000u) ? ~u : (u | 0x80000000u);
}
__device__ __forceinline__ float dec_f32(unsigned k) {
    unsigned u = (k & 0x80000000u) ? (k & 0x7FFFFFFFu) : ~k;
    return __uint_as_float(u);
}

// ---------------- init: h = 0, denom = 0, e_max = enc(-inf) ----------------
__global__ void init_kernel(float* __restrict__ h, float* __restrict__ denom,
                            unsigned* __restrict__ emax) {
    int i = blockIdx.x * blockDim.x + threadIdx.x;
    if (i < NN * DD) h[i] = 0.0f;
    if (i < NN) {
        denom[i] = 0.0f;
        emax[i]  = enc_f32(-__builtin_inff());   // = 0x007FFFFF
    }
}

// ------------- node kernel: z = logmap0(x); [s,t] = Z @ [w_src,w_dst] -------
// One wave handles a 16-node tile; 4 waves (128 threads) per block.
__global__ __launch_bounds__(128) void node_wmma_kernel(
        const float* __restrict__ x, const float* __restrict__ attn_w,
        float* __restrict__ z, float* __restrict__ s, float* __restrict__ t) {
    __shared__ float tile[4][16][DD];
    const int wave = threadIdx.x >> 5;
    const int lane = threadIdx.x & 31;
    const int m    = lane & 15;        // row within tile
    const int half = lane >> 4;        // which 32-dim half this lane owns
    const int tile_base = (blockIdx.x * 4 + wave) * 16;
    // NN is a multiple of 16, so a tile is either fully valid or fully OOB.
    const bool full = (tile_base + 16 <= NN);
    const int row  = tile_base + m;
    const int rowc = row < NN ? row : NN - 1;

    // logmap0: each lane processes 32 of the 64 dims of its row.
    const float* xr = x + (size_t)rowc * DD + half * 32;
    float v[32];
    float ss = 0.0f;
#pragma unroll
    for (int k = 0; k < 32; ++k) { v[k] = xr[k]; ss += v[k] * v[k]; }
    ss += __shfl_xor(ss, 16, 32);                       // pair with partner lane
    float pn    = fmaxf(sqrtf(ss), 1e-15f);
    float arg   = fminf(pn, 1.0f - 1e-7f);
    float scale = atanhf(arg) / pn;

    float* zt = &tile[wave][m][half * 32];
#pragma unroll
    for (int k = 0; k < 32; ++k) zt[k] = v[k] * scale;
    if (full) {                                        // uniform per-wave branch
        float* zg = z + (size_t)row * DD + half * 32;
#pragma unroll
        for (int k = 0; k < 32; ++k) zg[k] = zt[k];
    }
    __syncthreads();

    // Pack WMMA operands per ISA 16-bit A layout (and matching B layout):
    // lane L (L<16): element e holds K = e + (e>=8 ? 8 : 0)
    // lane L (L>=16): K shifted by +8.  Two K=32 blocks cover D=64.
    const int kbase = half * 8;
    v16h a0, a1, b0, b1;
#pragma unroll
    for (int e = 0; e < 16; ++e) {
        int K = kbase + e + ((e & 8) ? 8 : 0);
        a0[e] = (_Float16)tile[wave][m][K];
        a1[e] = (_Float16)tile[wave][m][32 + K];
        // B columns: n==0 -> w_src, n==1 -> w_dst, rest zero (n == m here).
        float w0 = (m == 0) ? attn_w[K]      : (m == 1) ? attn_w[DD + K]      : 0.0f;
        float w1 = (m == 0) ? attn_w[32 + K] : (m == 1) ? attn_w[DD + 32 + K] : 0.0f;
        b0[e] = (_Float16)w0;
        b1[e] = (_Float16)w1;
    }
    v8f c = {};
    c = __builtin_amdgcn_wmma_f32_16x16x32_f16(false, a0, false, b0, (short)0, c, false, false);
    c = __builtin_amdgcn_wmma_f32_16x16x32_f16(false, a1, false, b1, (short)0, c, false, false);

    // C/D layout: VGPR j, lanes 0-15 -> M=j, lanes 16-31 -> M=8+j; N = lane&15.
    // s lives in column N=0 (lanes 0,16); t in column N=1 (lanes 1,17).
    if (full && (lane & 14) == 0) {
        float* dp = ((lane & 1) ? t : s) + tile_base + half * 8;
#pragma unroll
        for (int j = 0; j < 8; ++j) dp[j] = c[j];      // straight-line stores
    }
}

// ---------------- edge pass 1: segment max of e over dst --------------------
__global__ void edge_max_kernel(const int* __restrict__ src, const int* __restrict__ dst,
                                const float* __restrict__ s, const float* __restrict__ t,
                                unsigned* __restrict__ emax) {
    int i = blockIdx.x * blockDim.x + threadIdx.x;
    if (i >= NE) return;
    float e = s[src[i]] + t[dst[i]];
    atomicMax(&emax[dst[i]], enc_f32(e));
}

// ---------------- edge pass 2: ex = exp(e - emax), denom += ex --------------
__global__ void edge_exp_kernel(const int* __restrict__ src, const int* __restrict__ dst,
                                const float* __restrict__ s, const float* __restrict__ t,
                                const unsigned* __restrict__ emax,
                                float* __restrict__ ex, float* __restrict__ denom) {
    int i = blockIdx.x * blockDim.x + threadIdx.x;
    if (i >= NE) return;
    int d   = dst[i];
    float e = s[src[i]] + t[d];
    float mx = dec_f32(emax[d]);
    if (!(fabsf(mx) < 3.4e38f)) mx = 0.0f;     // non-finite -> 0 (matches ref)
    float v = expf(e - mx);
    ex[i] = v;
    atomicAdd(&denom[d], v);
}

// ---------------- edge pass 3: h[dst] += alpha * z[src] ---------------------
// 16 threads per edge, 4 contiguous dims each (coalesced f32 atomics in L2).
__global__ void edge_agg_kernel(const int* __restrict__ src, const int* __restrict__ dst,
                                const float* __restrict__ ex, const float* __restrict__ denom,
                                const float* __restrict__ z, float* __restrict__ h) {
    int tid = blockIdx.x * blockDim.x + threadIdx.x;
    if (tid >= NE * 16) return;
    int edge = tid >> 4;
    int part = tid & 15;
    int d = dst[edge];
    int u = src[edge];
    float alpha = ex[edge] / denom[d];
    const float4 zv = *(const float4*)(z + (size_t)u * DD + part * 4);
    float* hp = h + (size_t)d * DD + part * 4;
    atomicAdd(hp + 0, alpha * zv.x);
    atomicAdd(hp + 1, alpha * zv.y);
    atomicAdd(hp + 2, alpha * zv.z);
    atomicAdd(hp + 3, alpha * zv.w);
}

// ---------------- final: out = proj(expmap0(h)) -----------------------------
__global__ __launch_bounds__(256) void final_kernel(const float* __restrict__ h,
                                                    float* __restrict__ out) {
    int wave = (blockIdx.x * blockDim.x + threadIdx.x) >> 5;
    int lane = threadIdx.x & 31;
    if (wave >= NN) return;
    const float* hp = h + (size_t)wave * DD + lane * 2;
    float2 v = *(const float2*)hp;
    float ss = v.x * v.x + v.y * v.y;
#pragma unroll
    for (int o = 16; o > 0; o >>= 1) ss += __shfl_xor(ss, o, 32);
    float normh = sqrtf(ss);
    float un    = fmaxf(normh, 1e-15f);
    float scale = tanhf(un) / un;                 // expmap0 factor
    float yx = v.x * scale, yy = v.y * scale;
    float ynorm = fmaxf(scale * normh, 1e-15f);   // ||y||
    const float maxnorm = 0.996f;                 // (1 - PROJ_EPS)/sqrt(C)
    if (ynorm > maxnorm) {
        float r = maxnorm / ynorm;
        yx *= r; yy *= r;
    }
    float* op = out + (size_t)wave * DD + lane * 2;
    op[0] = yx; op[1] = yy;
}

extern "C" void kernel_launch(void* const* d_in, const int* in_sizes, int n_in,
                              void* d_out, int out_size, void* d_ws, size_t ws_size,
                              hipStream_t stream) {
    const float* x      = (const float*)d_in[0];
    const int*   src    = (const int*)  d_in[1];
    const int*   dst    = (const int*)  d_in[2];
    const float* attn_w = (const float*)d_in[3];
    float*       out    = (float*)d_out;

    char* ws = (char*)d_ws;
    float*    z     = (float*)ws;                 ws += (size_t)NN * DD * sizeof(float);
    float*    h     = (float*)ws;                 ws += (size_t)NN * DD * sizeof(float);
    float*    s     = (float*)ws;                 ws += (size_t)NN * sizeof(float);
    float*    t     = (float*)ws;                 ws += (size_t)NN * sizeof(float);
    unsigned* emax  = (unsigned*)ws;              ws += (size_t)NN * sizeof(unsigned);
    float*    denom = (float*)ws;                 ws += (size_t)NN * sizeof(float);
    float*    ex    = (float*)ws;                 ws += (size_t)NE * sizeof(float);

    init_kernel<<<(NN * DD + 255) / 256, 256, 0, stream>>>(h, denom, emax);

    int node_blocks = (NN + 63) / 64;             // 4 waves * 16 nodes per block
    node_wmma_kernel<<<node_blocks, 128, 0, stream>>>(x, attn_w, z, s, t);

    edge_max_kernel<<<(NE + 255) / 256, 256, 0, stream>>>(src, dst, s, t, emax);
    edge_exp_kernel<<<(NE + 255) / 256, 256, 0, stream>>>(src, dst, s, t, emax, ex, denom);
    edge_agg_kernel<<<(NE * 16 + 255) / 256, 256, 0, stream>>>(src, dst, ex, denom, z, h);

    final_kernel<<<(NN * 32 + 255) / 256, 256, 0, stream>>>(h, out);
}
